// SelfAttentionLayer_87866440941704
// MI455X (gfx1250) — matrix operand order repaired and checked by reading, not compiled, not run on hardware
//
#include <hip/hip_runtime.h>
#include <hip/hip_bf16.h>

// ---------------------------------------------------------------------------
// Self-attention (B=8, T=2048, C=1024, H=128), fp32 in/out.
// f16 WMMA (v_wmma_f32_16x16x32_f16) + async global->LDS DMA, double-buffered.
//   Pass 0: transpose+convert W (C,H) f32 -> Wt (H,C) f16        (x3)
//   Pass 1: q,k = x@W  (f16, row-major T x H);  vT = (x@Wv)^T (f16, H x T)
//   Pass 2: flash attention with online softmax, causal mask, scale 1/sqrt(C)
// ---------------------------------------------------------------------------

typedef _Float16 v16h __attribute__((ext_vector_type(16)));
typedef _Float16 v8h  __attribute__((ext_vector_type(8)));
typedef float    v8f  __attribute__((ext_vector_type(8)));

#define WMMA16(a, b, c) \
    __builtin_amdgcn_wmma_f32_16x16x32_f16(false, (a), false, (b), (short)0, (c), false, false)

static constexpr int kT = 2048;
static constexpr int kC = 1024;
static constexpr int kH = 128;
static constexpr float kInvSqrtC = 0.03125f;     // 1/sqrt(1024)
static constexpr float kLog2e    = 1.44269504f;

__device__ __forceinline__ v16h cat8(v8h lo, v8h hi) {
    return __builtin_shufflevector(lo, hi, 0,1,2,3,4,5,6,7,8,9,10,11,12,13,14,15);
}

// --- CDNA5 async global->LDS DMA (ASYNCcnt) --------------------------------
__device__ __forceinline__ unsigned lds_off(const void* p) {
    // LDS aperture: low 32 bits of the generic address are the LDS byte offset
    return (unsigned)(size_t)p;
}
__device__ __forceinline__ void async_ld_b128(unsigned ldsoff, const void* g) {
    asm volatile("global_load_async_to_lds_b128 %0, %1, off"
                 :: "v"(ldsoff), "v"(g) : "memory");
}
__device__ __forceinline__ void wait_async0() {
    asm volatile("s_wait_asynccnt 0x0" ::: "memory");
}

// ---------------------------------------------------------------------------
// Pass 0: Wt[h*C + c] = (f16) W[c*H + h]   for Wq, Wk, Wv (blockIdx.y selects)
// ---------------------------------------------------------------------------
__global__ __launch_bounds__(128) void wt_convert_kernel(
    const float* __restrict__ Wq, const float* __restrict__ Wk, const float* __restrict__ Wv,
    _Float16* __restrict__ Wqt, _Float16* __restrict__ Wkt, _Float16* __restrict__ Wvt)
{
    const float* W = (blockIdx.y == 0) ? Wq : (blockIdx.y == 1) ? Wk : Wv;
    _Float16*   Wt = (blockIdx.y == 0) ? Wqt : (blockIdx.y == 1) ? Wkt : Wvt;
    int idx = blockIdx.x * 128 + threadIdx.x;            // over C*H = 131072
    if (idx < kC * kH) {
        int c = idx / kH;
        int h = idx % kH;
        Wt[(size_t)h * kC + c] = (_Float16)W[idx];
    }
}

// ---------------------------------------------------------------------------
// Pass 1: out = x @ W   (M = B*T = 16384 rows, K = C = 1024, N = H = 128)
// 128 threads = 4 waves; wave owns 16 rows x full N=128 (8 col blocks).
// Wt chunks (128 n x 32 c) double-buffered in LDS via async DMA.
// Output staged through LDS -> packed b128 global stores.
// TRANSPOSED==0: row-major f16 (B*T, H).  TRANSPOSED==1: vT layout (B, H, T).
// ---------------------------------------------------------------------------
template <int TRANSPOSED>
__global__ __launch_bounds__(128) void gemm_xw_kernel(
    const float* __restrict__ x, const _Float16* __restrict__ Wt,
    _Float16* __restrict__ outp)
{
    __shared__ __align__(128) _Float16 smem[8192];   // 2x 4096 W bufs; reused for out

    const int tid  = threadIdx.x;
    const int lane = tid & 31;
    const int wave = tid >> 5;
    const int ln   = lane & 15;       // A row / B-C column index
    const int hi   = lane >> 4;       // lane half

    const int mBase0 = blockIdx.x * 64;
    const int mBase  = mBase0 + wave * 16;
    const float* xrow = x + (size_t)(mBase + ln) * kC;

    v8f acc[8];
#pragma unroll
    for (int nb = 0; nb < 8; ++nb)
#pragma unroll
        for (int i = 0; i < 8; ++i) acc[nb][i] = 0.0f;

    // issue W chunk 0 into buf0: thread t loads 64 B of row n = t
    {
        const _Float16* src = Wt + (size_t)tid * kC;
        unsigned dst = lds_off(smem + tid * 32);
#pragma unroll
        for (int j = 0; j < 4; ++j) async_ld_b128(dst + j * 16, src + j * 8);
    }

    for (int kc = 0; kc < kC / 32; ++kc) {
        const int cur = kc & 1;
        wait_async0();
        __syncthreads();
        if (kc + 1 < kC / 32) {      // prefetch next W chunk into other buffer
            const _Float16* src = Wt + (size_t)tid * kC + (kc + 1) * 32;
            unsigned dst = lds_off(smem + (cur ^ 1) * 4096 + tid * 32);
#pragma unroll
            for (int j = 0; j < 4; ++j) async_ld_b128(dst + j * 16, src + j * 8);
        }

        // A operand: row ln, K runs at hi*8 and 16+hi*8 (ISA A-layout), cvt f32->f16
        const int c0 = kc * 32 + hi * 8;
        float4 f0 = *(const float4*)(xrow + c0);
        float4 f1 = *(const float4*)(xrow + c0 + 4);
        float4 f2 = *(const float4*)(xrow + c0 + 16);
        float4 f3 = *(const float4*)(xrow + c0 + 20);
        v16h a;
        a[0]=(_Float16)f0.x; a[1]=(_Float16)f0.y; a[2]=(_Float16)f0.z; a[3]=(_Float16)f0.w;
        a[4]=(_Float16)f1.x; a[5]=(_Float16)f1.y; a[6]=(_Float16)f1.z; a[7]=(_Float16)f1.w;
        a[8]=(_Float16)f2.x; a[9]=(_Float16)f2.y; a[10]=(_Float16)f2.z; a[11]=(_Float16)f2.w;
        a[12]=(_Float16)f3.x; a[13]=(_Float16)f3.y; a[14]=(_Float16)f3.z; a[15]=(_Float16)f3.w;

        // preload all 8 B operands, then back-to-back WMMAs
        const _Float16* wb = smem + cur * 4096;
        v16h bmat[8];
#pragma unroll
        for (int nb = 0; nb < 8; ++nb)
            bmat[nb] = cat8(*(const v8h*)(wb + (nb * 16 + ln) * 32 + hi * 8), *(const v8h*)(wb + (nb * 16 + ln) * 32 + 16 + hi * 8));
#pragma unroll
        for (int nb = 0; nb < 8; ++nb)
            acc[nb] = WMMA16(a, bmat[nb], acc[nb]);
    }

    // ---- epilogue: stage tile in LDS, then packed 128-bit global stores ----
    __syncthreads();
#pragma unroll
    for (int nb = 0; nb < 8; ++nb) {
#pragma unroll
        for (int r = 0; r < 8; ++r) {
            const int tl = wave * 16 + r + 8 * hi;   // local row 0..63
            const int n  = nb * 16 + ln;             // col 0..127
            const _Float16 vv = (_Float16)acc[nb][r];
            if (TRANSPOSED) smem[n * 64 + tl] = vv;  // [n][t] for vT
            else            smem[tl * 128 + n] = vv; // [t][n] row-major
        }
    }
    __syncthreads();
    if (TRANSPOSED) {
        // thread t stores head-row n = t: 64 f16 = 128 B
        const int bb = mBase0 >> 11;                  // / T
        const int tBase = mBase0 & (kT - 1);          // % T
        const uint4* s = (const uint4*)(smem + tid * 64);
        uint4* d = (uint4*)(outp + ((size_t)bb * kH + tid) * kT + tBase);
#pragma unroll
        for (int j = 0; j < 8; ++j) d[j] = s[j];
    } else {
        // thread t stores half a row: row = t>>1, 64 f16 = 128 B
        const int row = tid >> 1, half = tid & 1;
        const uint4* s = (const uint4*)(smem + row * 128 + half * 64);
        uint4* d = (uint4*)(outp + (size_t)(mBase0 + row) * kH + half * 64);
#pragma unroll
        for (int j = 0; j < 8; ++j) d[j] = s[j];
    }
}

// ---------------------------------------------------------------------------
// Pass 2: flash attention. 4 waves/WG; wave owns 16 queries; 32 keys/iter.
// K/V tiles double-buffered in LDS via async DMA.
// ---------------------------------------------------------------------------
__global__ __launch_bounds__(128) void flash_attn_kernel(
    const _Float16* __restrict__ q, const _Float16* __restrict__ k,
    const _Float16* __restrict__ vt, float* __restrict__ out)
{
    __shared__ __align__(128) _Float16 ldsK[2][32 * kH];      // [key][h]   2x8 KB
    __shared__ __align__(128) _Float16 ldsV[2][kH * 32];      // [h][key]   2x8 KB
    __shared__ __align__(128) _Float16 ldsP[4 * 16 * 32];     // per-wave P 4 KB

    const int tid  = threadIdx.x;
    const int lane = tid & 31;
    const int wave = tid >> 5;
    const int ln   = lane & 15;
    const int hi   = lane >> 4;

    const int b     = blockIdx.y;
    const int qtile = blockIdx.x * 64;
    const int qbase = qtile + wave * 16;
    const int qrowl = qbase + 8 * hi;          // query row for acc index r is qrowl + r

    const _Float16* kB = k  + (size_t)b * kT * kH;
    const _Float16* vB = vt + (size_t)b * kH * kT;

    // Q held in registers: 4 h-chunks in WMMA A-layout
    const _Float16* qrow = q + ((size_t)b * kT + qbase + ln) * kH;
    v16h qa[4];
#pragma unroll
    for (int c = 0; c < 4; ++c) {
        v8h lo = *(const v8h*)(qrow + 32 * c + hi * 16);
        v8h hh = *(const v8h*)(qrow + 32 * c + hi * 16 + 8);
        qa[c] = cat8(lo, hh);
    }

    v8f O[8];
    float m_[8], l_[8];
#pragma unroll
    for (int hb = 0; hb < 8; ++hb)
#pragma unroll
        for (int i = 0; i < 8; ++i) O[hb][i] = 0.0f;
#pragma unroll
    for (int r = 0; r < 8; ++r) { m_[r] = -1.0e30f; l_[r] = 0.0f; }

    _Float16* pw = ldsP + wave * 512;
    const int nkb = qtile / 32 + 2;            // key tiles needed (keys <= qtile+63)

    // issue tile 0
    {
        const int key = tid >> 2, qt = tid & 3;
        const _Float16* sk = kB + (size_t)key * kH + qt * 32;
        unsigned dk = lds_off(&ldsK[0][key * kH + qt * 32]);
        const _Float16* sv = vB + (size_t)tid * kT;
        unsigned dv = lds_off(&ldsV[0][tid * 32]);
#pragma unroll
        for (int j = 0; j < 4; ++j) { async_ld_b128(dk + j*16, sk + j*8);
                                      async_ld_b128(dv + j*16, sv + j*8); }
    }

    for (int kbi = 0; kbi < nkb; ++kbi) {
        const int kb  = kbi * 32;
        const int cur = kbi & 1;
        wait_async0();
        __syncthreads();
        if (kbi + 1 < nkb) {         // DMA next K/V tiles into the other buffers
            const int key = tid >> 2, qt = tid & 3;
            const _Float16* sk = kB + (size_t)(kb + 32 + key) * kH + qt * 32;
            unsigned dk = lds_off(&ldsK[cur ^ 1][key * kH + qt * 32]);
            const _Float16* sv = vB + (size_t)tid * kT + kb + 32;
            unsigned dv = lds_off(&ldsV[cur ^ 1][tid * 32]);
#pragma unroll
            for (int j = 0; j < 4; ++j) { async_ld_b128(dk + j*16, sk + j*8);
                                          async_ld_b128(dv + j*16, sv + j*8); }
        }

        // S = Q K^T: preload all 8 B operands, then 8 WMMAs back-to-back
        const _Float16* Kc = ldsK[cur];
        v16h bk[8];
#pragma unroll
        for (int c = 0; c < 4; ++c) {
            bk[c]     = *(const v16h*)(Kc + (ln)      * kH + 32 * c + hi * 16);
            bk[4 + c] = *(const v16h*)(Kc + (16 + ln) * kH + 32 * c + hi * 16);
        }
        v8f s0, s1;
#pragma unroll
        for (int i = 0; i < 8; ++i) { s0[i] = 0.0f; s1[i] = 0.0f; }
#pragma unroll
        for (int c = 0; c < 4; ++c) s0 = WMMA16(qa[c], bk[c], s0);
#pragma unroll
        for (int c = 0; c < 4; ++c) s1 = WMMA16(qa[c], bk[4 + c], s1);

        // online softmax (row stats per 16-lane half; xor masks 1/2/4/8)
        float sc[8];
#pragma unroll
        for (int r = 0; r < 8; ++r) {
            float a0 = s0[r] * kInvSqrtC;
            float a1 = s1[r] * kInvSqrtC;
            if (kb + ln      > qrowl + r) a0 = -1.0e30f;   // causal mask
            if (kb + 16 + ln > qrowl + r) a1 = -1.0e30f;
            float mt = fmaxf(a0, a1);
            mt = fmaxf(mt, __shfl_xor(mt, 1, 32));
            mt = fmaxf(mt, __shfl_xor(mt, 2, 32));
            mt = fmaxf(mt, __shfl_xor(mt, 4, 32));
            mt = fmaxf(mt, __shfl_xor(mt, 8, 32));
            const float mnew = fmaxf(m_[r], mt);
            const float scl  = exp2f((m_[r] - mnew) * kLog2e);
            const float p0   = exp2f((a0 - mnew) * kLog2e);
            const float p1   = exp2f((a1 - mnew) * kLog2e);
            pw[(r + 8 * hi) * 32 + ln]      = (_Float16)p0;   // transpose via LDS
            pw[(r + 8 * hi) * 32 + 16 + ln] = (_Float16)p1;
            float ps = p0 + p1;
            ps += __shfl_xor(ps, 1, 32);
            ps += __shfl_xor(ps, 2, 32);
            ps += __shfl_xor(ps, 4, 32);
            ps += __shfl_xor(ps, 8, 32);
            l_[r] = l_[r] * scl + ps;
            m_[r] = mnew;
            sc[r] = scl;
        }
#pragma unroll
        for (int hb = 0; hb < 8; ++hb)
#pragma unroll
            for (int r = 0; r < 8; ++r) O[hb][r] *= sc[r];

        __syncthreads();   // make P visible for the transposed A-layout read

        // P in A-layout + preloaded V operands, 8 WMMAs back-to-back
        v8h plo = *(const v8h*)(pw + ln * 32 + hi * 16);
        v8h phi = *(const v8h*)(pw + ln * 32 + hi * 16 + 8);
        v16h pa = cat8(plo, phi);
        const _Float16* Vc = ldsV[cur];
        v16h bv[8];
#pragma unroll
        for (int hb = 0; hb < 8; ++hb)
            bv[hb] = *(const v16h*)(Vc + (hb * 16 + ln) * 32 + hi * 16);
#pragma unroll
        for (int hb = 0; hb < 8; ++hb)
            O[hb] = WMMA16(pa, bv[hb], O[hb]);
    }

    // epilogue: out = O / l  (fp32, (B,T,H))
    float rcp[8];
#pragma unroll
    for (int r = 0; r < 8; ++r) rcp[r] = 1.0f / l_[r];
#pragma unroll
    for (int hb = 0; hb < 8; ++hb) {
#pragma unroll
        for (int r = 0; r < 8; ++r) {
            out[((size_t)b * kT + qrowl + r) * kH + hb * 16 + ln] = O[hb][r] * rcp[r];
        }
    }
}

// ---------------------------------------------------------------------------
extern "C" void kernel_launch(void* const* d_in, const int* in_sizes, int n_in,
                              void* d_out, int out_size, void* d_ws, size_t ws_size,
                              hipStream_t stream) {
    const float* x  = (const float*)d_in[0];   // (8, 2048, 1024)
    const float* Wq = (const float*)d_in[1];   // (1024, 128)
    const float* Wk = (const float*)d_in[2];
    const float* Wv = (const float*)d_in[3];
    float* out = (float*)d_out;                // (8, 2048, 128)

    const int M = 8 * kT;                      // 16384 rows
    _Float16* Wqt = (_Float16*)d_ws;           // 128*1024 each
    _Float16* Wkt = Wqt + (size_t)kH * kC;
    _Float16* Wvt = Wkt + (size_t)kH * kC;
    _Float16* qh  = Wvt + (size_t)kH * kC;     // (B*T, H) f16
    _Float16* kh  = qh  + (size_t)M * kH;      // (B*T, H) f16
    _Float16* vth = kh  + (size_t)M * kH;      // (B, H, T) f16

    wt_convert_kernel<<<dim3((kC * kH + 127) / 128, 3), 128, 0, stream>>>(
        Wq, Wk, Wv, Wqt, Wkt, Wvt);

    gemm_xw_kernel<0><<<M / 64, 128, 0, stream>>>(x, Wqt, qh);
    gemm_xw_kernel<0><<<M / 64, 128, 0, stream>>>(x, Wkt, kh);
    gemm_xw_kernel<1><<<M / 64, 128, 0, stream>>>(x, Wvt, vth);

    flash_attn_kernel<<<dim3(kT / 64, 8), 128, 0, stream>>>(qh, kh, vth, out);
}
